// MultiTaskModel_72069551227505
// MI455X (gfx1250) — compile-verified, matrix operand order checked by reading, not run on hardware
//
#include <hip/hip_runtime.h>
#include <hip/hip_bf16.h>
#include <math.h>

// ---------------------------------------------------------------------------
// MultiTaskModel forward for MI455X (gfx1250, wave32, WMMA + TDM).
// Heavy GEMMs + attention use v_wmma_f32_16x16x32_f16; A/Q tiles staged to
// LDS via the Tensor Data Mover (tensor_load_to_lds) when available.
// ---------------------------------------------------------------------------

#define B_   8
#define S_   1024
#define CIN_ 16
#define E_   512
#define H_   8
#define HD_  64
#define L_   4
#define FFN_ 2048
#define EPS_ 1e-5f

typedef __attribute__((ext_vector_type(16))) _Float16 v16h;
typedef __attribute__((ext_vector_type(8)))  float    v8f;
typedef __attribute__((ext_vector_type(4)))  unsigned v4u;
typedef __attribute__((ext_vector_type(4)))  int      v4i;
typedef __attribute__((ext_vector_type(8)))  int      v8i;

#if __has_include(<hip/amd_detail/amd_gfx1250_TDM.h>)
#define TDM_6ARG 1
#endif

__device__ __forceinline__ float gelu_exact(float x) {
  return 0.5f * x * (1.f + erff(x * 0.70710678118654752f));
}

// ---------------------------------------------------------------------------
// Tensor Data Mover: 2D tile load global(f32)->LDS with TDM-side row padding.
// D# layout per CDNA5 ISA 8.3-8.6.  pad: after (1<<pad_interval_code) DWORD
// pairs*... -> we use interval code 4 (32 dw) / 5 (64 dw), amount code 0 (1 dw)
// so a 32-float row lands on a 33-float LDS stride (65 for 64-float rows).
// ---------------------------------------------------------------------------
#if __has_builtin(__builtin_amdgcn_tensor_load_to_lds)
#define HAVE_TDM 1
__device__ __forceinline__ void tdm_load_2d_f32(unsigned lds_off, const float* gptr,
                                                unsigned tile_d0, unsigned tile_d1,
                                                unsigned long long stride_elems,
                                                unsigned tensor_d0, unsigned tensor_d1,
                                                unsigned pad_interval_code)
{
  unsigned long long ga = (unsigned long long)(const void*)gptr;
  v4u g0;
  g0.x = 1u;                                             // count=1 (valid user D#)
  g0.y = lds_off;                                        // lds_addr (bytes)
  g0.z = (unsigned)(ga & 0xFFFFFFFFull);                 // global_addr[31:0]
  g0.w = (unsigned)((ga >> 32) & 0x01FFFFFFull)          // global_addr[56:32]
       | 0x80000000u;                                    // type = 2 ("image")
  v8i g1;
  g1[0] = (int)((2u << 16)                               // data_size = 4B
              | (1u << 20)                               // pad_enable
              | (pad_interval_code << 22));              // pad_interval, amount=0(1dw)
  g1[1] = (int)((tensor_d0 & 0xFFFFu) << 16);            // tensor_dim0[15:0]
  g1[2] = (int)((tensor_d0 >> 16) | ((tensor_d1 & 0xFFFFu) << 16));
  g1[3] = (int)((tensor_d1 >> 16) | (tile_d0 << 16));    // tile_dim0
  g1[4] = (int)(tile_d1 & 0xFFFFu);                      // tile_dim1 (tile_dim2=0)
  g1[5] = (int)(stride_elems & 0xFFFFFFFFull);           // tensor_dim0_stride lo
  g1[6] = (int)((stride_elems >> 32) & 0xFFFFull);       // stride hi (dim1_stride=0)
  g1[7] = 0;
  v4i gz = {0, 0, 0, 0};
#ifdef TDM_6ARG
  v8i gz8 = {0, 0, 0, 0, 0, 0, 0, 0};
  __builtin_amdgcn_tensor_load_to_lds(g0, g1, gz, gz, gz8, 0);
#else
  __builtin_amdgcn_tensor_load_to_lds(g0, g1, gz, gz, 0);
#endif
}
#endif

// ---------------------------------------------------------------------------
// Conv bank (k=3,5,7,9) + BN + GELU + sinusoidal PE + pos_bias -> h
// ---------------------------------------------------------------------------
struct ConvP { const float* W; const float* b; const float* g; const float* be;
               const float* rm; const float* rv; };
struct ConvArgs { ConvP c[4]; };

__global__ void conv_pe_kernel(const float* __restrict__ x, ConvArgs cp,
                               const float* __restrict__ pos_bias,
                               float* __restrict__ h)
{
  int idx = blockIdx.x * 256 + threadIdx.x;      // B*S*E threads exactly
  int e  = idx & (E_ - 1);
  int s  = (idx >> 9) & (S_ - 1);
  int bb = idx >> 19;
  int g  = e >> 7;
  int lc = e & 127;
  int ksz = 3 + 2 * g, half = ksz >> 1;
  const float* wp = cp.c[g].W + lc * CIN_ * ksz;
  const float* xb = x + (long)bb * S_ * CIN_;
  float sum = 0.f;
  for (int dt = 0; dt < ksz; ++dt) {
    int ss = s + dt - half;
    if (ss >= 0 && ss < S_) {
      const float* xr = xb + ss * CIN_;
      #pragma unroll
      for (int c = 0; c < CIN_; ++c) sum += xr[c] * wp[c * ksz + dt];
    }
  }
  sum += cp.c[g].b[lc];
  sum = (sum - cp.c[g].rm[lc]) * rsqrtf(cp.c[g].rv[lc] + EPS_);
  sum = sum * cp.c[g].g[lc] + cp.c[g].be[lc];
  sum = gelu_exact(sum);
  int j2 = e & ~1;
  float divj = expf((float)j2 * (-9.210340371976184f / (float)E_));
  float ang  = (float)s * divj;
  float pe   = (e & 1) ? cosf(ang) : sinf(ang);
  h[idx] = sum + pe + pos_bias[s * E_ + e];
}

// ---------------------------------------------------------------------------
// LayerNorm: one wave per row of 512, wave32 shfl reductions.
// ---------------------------------------------------------------------------
__global__ void ln_kernel(const float* __restrict__ in,
                          const float* __restrict__ g,
                          const float* __restrict__ bta,
                          float* __restrict__ out)
{
  int wave = threadIdx.x >> 5, lane = threadIdx.x & 31;
  int row = blockIdx.x * 8 + wave;
  const float* r = in + (long)row * E_;
  float s = 0.f, s2 = 0.f;
  #pragma unroll
  for (int j = 0; j < E_ / 32; ++j) {
    float v = r[lane + 32 * j];
    s += v; s2 += v * v;
  }
  #pragma unroll
  for (int o = 16; o >= 1; o >>= 1) {
    s  += __shfl_xor(s,  o, 32);
    s2 += __shfl_xor(s2, o, 32);
  }
  float mu  = s  * (1.f / E_);
  float var = s2 * (1.f / E_) - mu * mu;
  float inv = rsqrtf(var + EPS_);
  #pragma unroll
  for (int j = 0; j < E_ / 32; ++j) {
    int e = lane + 32 * j;
    out[(long)row * E_ + e] = (r[e] - mu) * inv * g[e] + bta[e];
  }
}

// ---------------------------------------------------------------------------
// Gate scalars: qg = q @ Wg[:E], kg = k @ Wg[E:].  One wave per row.
// ---------------------------------------------------------------------------
__global__ void gate_kernel(const float* __restrict__ q,
                            const float* __restrict__ k,
                            const float* __restrict__ Wg,
                            float* __restrict__ qg, float* __restrict__ kg)
{
  int wave = threadIdx.x >> 5, lane = threadIdx.x & 31;
  int row = blockIdx.x * 8 + wave;
  float a = 0.f, b = 0.f;
  #pragma unroll
  for (int j = 0; j < E_ / 32; ++j) {
    int e = lane + 32 * j;
    a += q[(long)row * E_ + e] * Wg[e];
    b += k[(long)row * E_ + e] * Wg[E_ + e];
  }
  #pragma unroll
  for (int o = 16; o >= 1; o >>= 1) {
    a += __shfl_xor(a, o, 32);
    b += __shfl_xor(b, o, 32);
  }
  if (lane == 0) { qg[row] = a; kg[row] = b; }
}

// ---------------------------------------------------------------------------
// WMMA GEMM: out[M,N] = act(A[M,K](lda) * W[K,N] + bias) (+res).
// 256 threads = 8 waves (4x2 grid of 16x16 tiles), BM=64 BN=32, K-step 32.
// A tile staged f32 via TDM (33-float padded rows); W tile staged f16
// transposed by the threads.  Fragments follow CDNA5 16-bit A/B layout.
// ---------------------------------------------------------------------------
template<bool GELU_ACT, bool RES>
__global__ void gemm_wmma(const float* __restrict__ A, long lda,
                          const float* __restrict__ W,
                          const float* __restrict__ bias,
                          const float* res, float* out,
                          int M, int N, int K)
{
  __shared__ float    Af[64 * 33];   // A tile f32, TDM-padded stride 33
  __shared__ _Float16 Bl[32 * 40];   // W tile transposed [n][k], stride 40
  const int tid  = threadIdx.x;
  const int lane = tid & 31;
  const int wv   = tid >> 5;
  const int wm   = wv >> 1;
  const int wn   = wv & 1;
  const int mBase = blockIdx.y * 64;
  const int nBase = blockIdx.x * 32;
  const int l16 = lane & 15;
  const int cb  = (lane < 16) ? 0 : 8;
  const int mValid = (M - mBase < 64) ? (M - mBase) : 64;

  v8f acc = {};
  for (int k0 = 0; k0 < K; k0 += 32) {
#ifdef HAVE_TDM
    if (wv == 0) {
      tdm_load_2d_f32((unsigned)(unsigned long long)(void*)Af,
                      A + (long)mBase * lda + k0,
                      /*tile*/ 32, 64, (unsigned long long)lda,
                      /*tensor*/ 32, (unsigned)mValid,
                      /*pad interval code: 32 DWORDs*/ 4);
    }
#else
    #pragma unroll
    for (int j = 0; j < 8; ++j) {
      int e = tid + 256 * j;
      int r = e >> 5, c = e & 31;
      int gm = mBase + r;
      Af[r * 33 + c] = (gm < M) ? A[(long)gm * lda + k0 + c] : 0.f;
    }
#endif
    // cooperative W load (32x32 f32 -> f16), stored transposed [n][k]
    #pragma unroll
    for (int j = 0; j < 4; ++j) {
      int e = tid + 256 * j;
      int r = e >> 5, c = e & 31;    // r = k-in-tile, c = n-in-tile
      int gn = nBase + c;
      float v = (gn < N) ? W[(long)(k0 + r) * N + gn] : 0.f;
      Bl[c * 40 + r] = (_Float16)v;
    }
    if (k0 + 32 < K) {  // prefetch next weight tile -> global_prefetch_b8
      __builtin_prefetch(&W[(long)(k0 + 32 + (tid >> 3)) * N + nBase + (tid & 7) * 4], 0, 1);
    }
#ifdef HAVE_TDM
#if __has_builtin(__builtin_amdgcn_s_wait_tensorcnt)
    if (wv == 0) __builtin_amdgcn_s_wait_tensorcnt((short)0);
#endif
#endif
    __syncthreads();
    v16h af, bf;
    const float*    arow = Af + (wm * 16 + l16) * 33;
    const _Float16* brow = Bl + (wn * 16 + l16) * 40;
    #pragma unroll
    for (int i = 0; i < 8; ++i) {
      af[i]     = (_Float16)arow[cb + i];
      af[8 + i] = (_Float16)arow[cb + 16 + i];
      bf[i]     = brow[cb + i];
      bf[8 + i] = brow[cb + 16 + i];
    }
    acc = __builtin_amdgcn_wmma_f32_16x16x32_f16(false, af, false, bf,
                                                 (short)0, acc, false, false);
    __syncthreads();
  }
  // epilogue (C/D layout: lane -> n = l16; vgpr r -> m = r (+8 lanes 16..31))
  const int n    = nBase + wn * 16 + l16;
  const int mTop = mBase + wm * 16 + ((lane < 16) ? 0 : 8);
  if (n < N) {
    float bv = bias ? bias[n] : 0.f;
    #pragma unroll
    for (int r = 0; r < 8; ++r) {
      int m = mTop + r;
      if (m < M) {
        float v = acc[r] + bv;
        if (GELU_ACT) v = gelu_exact(v);
        if (RES)      v += res[(long)m * N + n];
        out[(long)m * N + n] = v;
      }
    }
  }
}

// ---------------------------------------------------------------------------
// Fused windowed-gated attention. One block per (b, head, 16-row q tile).
// LDS: Q tile f32 (TDM, 65-float padded rows) + 16x1024 f32 score strip
// (64KB -- needs CDNA5's 320KB/WGP LDS) + AV partial-sum buffer.
// ---------------------------------------------------------------------------
#define ATTN_SMEM (16*65*4 + 16*1024*4 + 8*256*4)   // 4160+65536+8192 = 77888

__global__ void attn_kernel(const float* __restrict__ q,
                            const float* __restrict__ k,
                            const float* __restrict__ v,
                            const float* __restrict__ qg,
                            const float* __restrict__ kg,
                            const float* __restrict__ gateb,
                            float* __restrict__ ao)
{
  extern __shared__ char smem[];
  float* Qf  = (float*)smem;                       // 16 x 65
  float* Ss  = (float*)(smem + 16 * 65 * 4);       // 16 x 1024 scores
  float* Pac = Ss + 16 * 1024;                     // 8 waves x 256 partials

  const int qt = blockIdx.x, hh = blockIdx.y, bb = blockIdx.z;
  const int qBase = qt * 16;
  const int hBase = hh * HD_;
  const int tid = threadIdx.x, lane = tid & 31, wv = tid >> 5;
  const int l16 = lane & 15;
  const int cb  = (lane < 16) ? 0 : 8;
  const float bg = gateb[0];

  // stage Q tile (16 rows x 64 feat) -> LDS
#ifdef HAVE_TDM
  if (wv == 0) {
    tdm_load_2d_f32((unsigned)(unsigned long long)(void*)Qf,
                    q + ((long)(bb * S_ + qBase)) * E_ + hBase,
                    /*tile*/ 64, 16, (unsigned long long)E_,
                    /*tensor*/ 64, 16,
                    /*pad interval code: 64 DWORDs*/ 5);
#if __has_builtin(__builtin_amdgcn_s_wait_tensorcnt)
    __builtin_amdgcn_s_wait_tensorcnt((short)0);
#endif
  }
#else
  #pragma unroll
  for (int j = 0; j < 4; ++j) {
    int e = tid + 256 * j;
    int r = e >> 6, c = e & 63;
    Qf[r * 65 + c] = q[((long)(bb * S_ + qBase + r)) * E_ + hBase + c];
  }
#endif
  __syncthreads();

  // ---- Pass A: scores = (Q K^T)*scale + gauss bias, gated + windowed -----
  for (int kt = wv; kt < S_ / 16; kt += 8) {
    int kpos = kt * 16 + l16;
    v8f acc = {};
    #pragma unroll
    for (int kk = 0; kk < HD_; kk += 32) {
      v16h af, bf;
      const float* kp = k + ((long)(bb * S_ + kpos)) * E_ + hBase + kk;
      #pragma unroll
      for (int i = 0; i < 8; ++i) {
        af[i]     = (_Float16)Qf[l16 * 65 + kk + cb + i];
        af[8 + i] = (_Float16)Qf[l16 * 65 + kk + cb + 16 + i];
        bf[i]     = (_Float16)kp[cb + i];
        bf[8 + i] = (_Float16)kp[cb + 16 + i];
      }
      acc = __builtin_amdgcn_wmma_f32_16x16x32_f16(false, af, false, bf,
                                                   (short)0, acc, false, false);
    }
    float kgv = kg[bb * S_ + kpos];
    #pragma unroll
    for (int r = 0; r < 8; ++r) {
      int m = r + ((lane < 16) ? 0 : 8);
      int qpos = qBase + m;
      float d  = fabsf((float)(qpos - kpos));
      float sc = acc[r] * 0.125f - d * d * (1.f / 131072.f);
      float gl = qg[bb * S_ + qpos] + kgv + bg;
      float gate = 1.f / (1.f + expf(-gl));
      float s = sc * gate;
      if (d > 512.f) s = -__builtin_inff();
      Ss[m * S_ + kpos] = s;
    }
  }
  __syncthreads();

  // ---- Pass B: row softmax (2 rows per wave) -----------------------------
  for (int m = wv * 2; m < wv * 2 + 2; ++m) {
    float mx = -__builtin_inff();
    #pragma unroll 4
    for (int j = 0; j < S_ / 32; ++j) mx = fmaxf(mx, Ss[m * S_ + lane + 32 * j]);
    #pragma unroll
    for (int o = 16; o >= 1; o >>= 1) mx = fmaxf(mx, __shfl_xor(mx, o, 32));
    float sum = 0.f;
    #pragma unroll 4
    for (int j = 0; j < S_ / 32; ++j) {
      int idx = m * S_ + lane + 32 * j;
      float p = expf(Ss[idx] - mx);
      Ss[idx] = p; sum += p;
    }
    #pragma unroll
    for (int o = 16; o >= 1; o >>= 1) sum += __shfl_xor(sum, o, 32);
    float inv = 1.f / sum;
    #pragma unroll 4
    for (int j = 0; j < S_ / 32; ++j) Ss[m * S_ + lane + 32 * j] *= inv;
  }
  __syncthreads();

  // ---- Pass C: O = P V.  8 waves = 4 N-tiles x 2 K-halves ----------------
  const int nt = wv & 3, kh = wv >> 2;
  v8f acc = {};
  const int f = hBase + nt * 16 + l16;
  for (int ks = kh * 512; ks < kh * 512 + 512; ks += 32) {
    v16h af, bf;
    #pragma unroll
    for (int i = 0; i < 8; ++i) {
      af[i]     = (_Float16)Ss[l16 * S_ + ks + cb + i];
      af[8 + i] = (_Float16)Ss[l16 * S_ + ks + cb + 16 + i];
      bf[i]     = (_Float16)v[((long)(bb * S_ + ks + cb + i)) * E_ + f];
      bf[8 + i] = (_Float16)v[((long)(bb * S_ + ks + cb + 16 + i)) * E_ + f];
    }
    acc = __builtin_amdgcn_wmma_f32_16x16x32_f16(false, af, false, bf,
                                                 (short)0, acc, false, false);
  }
  #pragma unroll
  for (int r = 0; r < 8; ++r) Pac[wv * 256 + r * 32 + lane] = acc[r];
  __syncthreads();
  if (wv < 4) {
    #pragma unroll
    for (int r = 0; r < 8; ++r) {
      float s = Pac[wv * 256 + r * 32 + lane] + Pac[(wv + 4) * 256 + r * 32 + lane];
      int m = r + ((lane < 16) ? 0 : 8);
      ao[((long)(bb * S_ + qBase + m)) * E_ + hBase + nt * 16 + l16] = s;
    }
  }
#if __has_builtin(__builtin_amdgcn_s_cluster_barrier)
  __builtin_amdgcn_s_cluster_barrier();   // NOP when not in a cluster (ISA 6.6)
#endif
}

// ---------------------------------------------------------------------------
// Host-side orchestration (graph-capture-safe: only stream launches).
// ---------------------------------------------------------------------------
extern "C" void kernel_launch(void* const* d_in, const int* in_sizes, int n_in,
                              void* d_out, int out_size, void* d_ws, size_t ws_size,
                              hipStream_t stream)
{
  (void)in_sizes; (void)n_in; (void)out_size; (void)ws_size;
  const float* x = (const float*)d_in[0];
  ConvArgs ca;
  for (int i = 0; i < 4; ++i) {
    int b = 1 + 6 * i;
    ca.c[i].W  = (const float*)d_in[b + 0];
    ca.c[i].b  = (const float*)d_in[b + 1];
    ca.c[i].g  = (const float*)d_in[b + 2];
    ca.c[i].be = (const float*)d_in[b + 3];
    ca.c[i].rm = (const float*)d_in[b + 4];
    ca.c[i].rv = (const float*)d_in[b + 5];
  }
  const float* pos_bias = (const float*)d_in[25];
  const float* clsW = (const float*)d_in[98];
  const float* clsb = (const float*)d_in[99];
  const float* r1W  = (const float*)d_in[100];
  const float* r1b  = (const float*)d_in[101];
  const float* r2W  = (const float*)d_in[102];
  const float* r2b  = (const float*)d_in[103];

  // workspace carve (floats)
  const long T = (long)B_ * S_ * E_;            // 4,194,304
  float* h   = (float*)d_ws;
  float* hn  = h  + T;
  float* q   = hn + T;
  float* kx  = q  + T;
  float* v   = kx + T;
  float* x1  = v  + T;                          // spillover for FFN scratch
  float* t   = q;                               // FFN scratch spans q..x1
  float* ao  = hn;                              // attention out reuses hn
  float* qg  = x1 + T;
  float* kg  = qg + (long)B_ * S_;
  float* rt  = kg + (long)B_ * S_;              // reg-head intermediate (8x512)

  const int ROWS = B_ * S_;                     // 8192
  dim3 gEE(E_ / 32, ROWS / 64);
  dim3 gEF(FFN_ / 32, ROWS / 64);

  conv_pe_kernel<<<(B_ * S_ * E_) / 256, 256, 0, stream>>>(x, ca, pos_bias, h);

  for (int l = 0; l < L_; ++l) {
    int base = 26 + 18 * l;
    const float* Wq = (const float*)d_in[base + 0];
    const float* bq = (const float*)d_in[base + 1];
    const float* Wk = (const float*)d_in[base + 2];
    const float* bk = (const float*)d_in[base + 3];
    const float* Wv = (const float*)d_in[base + 4];
    const float* bv = (const float*)d_in[base + 5];
    const float* Wo = (const float*)d_in[base + 6];
    const float* bo = (const float*)d_in[base + 7];
    const float* Wg = (const float*)d_in[base + 8];
    const float* gb = (const float*)d_in[base + 9];
    const float* l1g = (const float*)d_in[base + 10];
    const float* l1b = (const float*)d_in[base + 11];
    const float* l2g = (const float*)d_in[base + 12];
    const float* l2b = (const float*)d_in[base + 13];
    const float* W1 = (const float*)d_in[base + 14];
    const float* b1 = (const float*)d_in[base + 15];
    const float* W2 = (const float*)d_in[base + 16];
    const float* b2 = (const float*)d_in[base + 17];

    ln_kernel<<<ROWS / 8, 256, 0, stream>>>(h, l1g, l1b, hn);
    gemm_wmma<false, false><<<gEE, 256, 0, stream>>>(hn, E_, Wq, bq, nullptr, q,  ROWS, E_, E_);
    gemm_wmma<false, false><<<gEE, 256, 0, stream>>>(hn, E_, Wk, bk, nullptr, kx, ROWS, E_, E_);
    gemm_wmma<false, false><<<gEE, 256, 0, stream>>>(hn, E_, Wv, bv, nullptr, v,  ROWS, E_, E_);
    gate_kernel<<<ROWS / 8, 256, 0, stream>>>(q, kx, Wg, qg, kg);
    attn_kernel<<<dim3(S_ / 16, H_, B_), 256, ATTN_SMEM, stream>>>(q, kx, v, qg, kg, gb, ao);
    gemm_wmma<false, true ><<<gEE, 256, 0, stream>>>(ao, E_, Wo, bo, h, h, ROWS, E_, E_);
    ln_kernel<<<ROWS / 8, 256, 0, stream>>>(h, l2g, l2b, hn);
    gemm_wmma<true,  false><<<gEF, 256, 0, stream>>>(hn, E_, W1, b1, nullptr, t, ROWS, FFN_, E_);
    gemm_wmma<false, true ><<<gEE, 256, 0, stream>>>(t, FFN_, W2, b2, h, h, ROWS, E_, FFN_);
  }

  // heads on feat = h[:, -1]  (row stride S*E)
  const float* feat = h + (long)(S_ - 1) * E_;
  const long featStride = (long)S_ * E_;
  float* cls_out = (float*)d_out;          // 8 x 10
  float* reg_out = cls_out + B_ * 10;      // 8 x 1024
  gemm_wmma<false, false><<<dim3(1, 1),  256, 0, stream>>>(feat, featStride, clsW, clsb, nullptr, cls_out, B_, 10,   E_);
  gemm_wmma<true,  false><<<dim3(16, 1), 256, 0, stream>>>(feat, featStride, r1W,  r1b,  nullptr, rt,      B_, E_,   E_);
  gemm_wmma<false, false><<<dim3(32, 1), 256, 0, stream>>>(rt,   E_,         r2W,  r2b,  nullptr, reg_out, B_, 1024, E_);
}
